// MultiHeadAttention_85753317032258
// MI455X (gfx1250) — matrix operand order repaired and checked by reading, not compiled, not run on hardware
//
#include <hip/hip_runtime.h>

// ---------------------------------------------------------------------------
// MI455X (gfx1250) fused multi-head attention, bf16 WMMA pipeline.
// B=2, T=2048, C=1024, H=16, HD=64.
// ---------------------------------------------------------------------------

#define Bc 2
#define Tc 2048
#define Cc 1024
#define Hc 16
#define HDc 64
#define ROWS (Bc * Tc)          // 4096

typedef __attribute__((ext_vector_type(8)))  __bf16 v8bf;
typedef __attribute__((ext_vector_type(16))) __bf16 v16bf;
typedef __attribute__((ext_vector_type(8)))  float  v8f;

union V16U { v8bf h[2]; v16bf v; };

__device__ __forceinline__ unsigned short f2bf(float f) {
    unsigned u = __float_as_uint(f);
    unsigned r = u + 0x7FFFu + ((u >> 16) & 1u);   // round-to-nearest-even
    return (unsigned short)(r >> 16);
}

// A-fragment (16x32, 16-bit): lane L holds row M=L&15; K chunks per ISA table:
// lanes 0-15 -> K = k0+[0..7], k0+[16..23]; lanes 16-31 -> +8.
__device__ __forceinline__ v16bf load_frag_A(const unsigned short* base, int row0,
                                             int ld, int k0, int lane) {
    int m  = lane & 15;
    int hi = lane >> 4;
    const unsigned short* p = base + (size_t)(row0 + m) * ld + k0 + hi * 8;
    V16U u;
    u.h[0] = *reinterpret_cast<const v8bf*>(p);
    u.h[1] = *reinterpret_cast<const v8bf*>(p + 16);
    return u.v;
}

// B-fragment (32x16, 16-bit): lane L holds column N=L&15; lanes 0-15 cover
// K=k0+[0..7], k0+[16..23]; lanes 16-31 -> +8 (matches A-fragment order).
// `base` is the transposed (N-major, K-contiguous) matrix.
__device__ __forceinline__ v16bf load_frag_B(const unsigned short* base, int n0,
                                             int ld, int k0, int lane) {
    int n  = lane & 15;
    int hi = lane >> 4;
    const unsigned short* p = base + (size_t)(n0 + n) * ld + k0 + hi * 8;
    V16U u;
    u.h[0] = *reinterpret_cast<const v8bf*>(p);
    u.h[1] = *reinterpret_cast<const v8bf*>(p + 16);
    return u.v;
}

#define WMMA_BF16(a, b, c) \
    __builtin_amdgcn_wmma_f32_16x16x32_bf16(false, (a), false, (b), (short)0, (c), false, false)

// ---------------------------------------------------------------------------
// Conversion / layout kernels
// ---------------------------------------------------------------------------
__global__ void cvt_bf16_kernel(const float* __restrict__ in,
                                unsigned short* __restrict__ out, int n) {
    int i = blockIdx.x * blockDim.x + threadIdx.x;
    if (i < n) out[i] = f2bf(in[i]);
}

// [H][C][HD] f32 -> [H][HD][C] bf16
__global__ void cvt_wT_kernel(const float* __restrict__ in,
                              unsigned short* __restrict__ out) {
    int i = blockIdx.x * blockDim.x + threadIdx.x;   // H*C*HD = 1048576
    int h = i >> 16;                                 // C*HD = 65536
    int r = i & 65535;
    int c = r >> 6;
    int d = r & 63;
    out[((size_t)h * HDc + d) * Cc + c] = f2bf(in[i]);
}

// Wp [C][C] f32 (row=j, col=cO) -> WpT [C][C] bf16 (row=cO, col=j)
__global__ void cvt_wpT_kernel(const float* __restrict__ in,
                               unsigned short* __restrict__ out) {
    int i = blockIdx.x * blockDim.x + threadIdx.x;   // 1048576
    int j  = i >> 10;
    int co = i & 1023;
    out[(size_t)co * Cc + j] = f2bf(in[i]);
}

// ---------------------------------------------------------------------------
// QKV projection: per wave, 16 rows x 64 cols (full head), K-loop over C.
// Q scaled by HD^-0.5. V stored transposed [B,H,HD,T].
// ---------------------------------------------------------------------------
__global__ __launch_bounds__(128) void qkv_kernel(
    const unsigned short* __restrict__ xb,
    const unsigned short* __restrict__ WqT,
    const unsigned short* __restrict__ WkT,
    const unsigned short* __restrict__ WvT,
    unsigned short* __restrict__ Q,
    unsigned short* __restrict__ K,
    unsigned short* __restrict__ VT) {
    int lane = threadIdx.x & 31;
    int w    = blockIdx.x * 4 + (threadIdx.x >> 5);   // 0..4095
    int h    = w >> 8;                                // head
    int mt   = w & 255;
    int row0 = mt * 16;                               // row in [0,4096)

    v8f accQ[4], accK[4], accV[4];
    v8f z = {0.f, 0.f, 0.f, 0.f, 0.f, 0.f, 0.f, 0.f};
    for (int j = 0; j < 4; ++j) { accQ[j] = z; accK[j] = z; accV[j] = z; }

    const unsigned short* wq = WqT + (size_t)h * HDc * Cc;
    const unsigned short* wk = WkT + (size_t)h * HDc * Cc;
    const unsigned short* wv = WvT + (size_t)h * HDc * Cc;

    for (int k0 = 0; k0 < Cc; k0 += 32) {
        v16bf a = load_frag_A(xb, row0, Cc, k0, lane);
        for (int j = 0; j < 4; ++j) {
            v16bf bq = load_frag_B(wq, j * 16, Cc, k0, lane);
            accQ[j] = WMMA_BF16(a, bq, accQ[j]);
            v16bf bk = load_frag_B(wk, j * 16, Cc, k0, lane);
            accK[j] = WMMA_BF16(a, bk, accK[j]);
            v16bf bv = load_frag_B(wv, j * 16, Cc, k0, lane);
            accV[j] = WMMA_BF16(a, bv, accV[j]);
        }
    }

    int hi = lane >> 4, nlo = lane & 15;
    int b  = row0 >> 11;
    int t0 = row0 & (Tc - 1);
    size_t qkBase = (size_t)(b * Hc + h) * Tc * HDc;
    size_t vtBase = (size_t)(b * Hc + h) * HDc * Tc;
    const float qscale = 0.125f;   // HD^-0.5

    for (int j = 0; j < 4; ++j) {
        for (int v = 0; v < 8; ++v) {
            int t = t0 + v + 8 * hi;
            int d = j * 16 + nlo;
            Q[qkBase + (size_t)t * HDc + d] = f2bf(accQ[j][v] * qscale);
            K[qkBase + (size_t)t * HDc + d] = f2bf(accK[j][v]);
            VT[vtBase + (size_t)d * Tc + t] = f2bf(accV[j][v]);
        }
    }
}

// ---------------------------------------------------------------------------
// Flash attention: one wave owns 16 query rows, streams causal s-chunks of 32.
// D-fragment row m lives at vgpr v=m%8, lanes (m<8 ? 0-15 : 16-31), so per-row
// softmax stats are per-lane float[8] + 16-lane half-wave shuffles.
// ---------------------------------------------------------------------------
__global__ __launch_bounds__(128) void flash_kernel(
    const unsigned short* __restrict__ Q,
    const unsigned short* __restrict__ K,
    const unsigned short* __restrict__ VT,
    unsigned short* __restrict__ attn) {
    __shared__ __align__(16) unsigned short pbuf[4][16 * 32];

    int lane = threadIdx.x & 31;
    int wave = threadIdx.x >> 5;
    int w    = blockIdx.x * 4 + wave;   // 0..4095
    int bh   = w >> 7;                  // 0..31  (b*16+h)
    int mt   = w & 127;
    int row0 = mt * 16;                 // t0
    int hi = lane >> 4, nlo = lane & 15;

    const unsigned short* qb = Q  + (size_t)bh * Tc * HDc;
    const unsigned short* kb = K  + (size_t)bh * Tc * HDc;
    const unsigned short* vb = VT + (size_t)bh * HDc * Tc;
    unsigned short* lds = pbuf[wave];

    v16bf qf0 = load_frag_A(qb, row0, HDc, 0, lane);
    v16bf qf1 = load_frag_A(qb, row0, HDc, 32, lane);

    v8f z = {0.f, 0.f, 0.f, 0.f, 0.f, 0.f, 0.f, 0.f};
    v8f O[4];
    for (int j = 0; j < 4; ++j) O[j] = z;
    float mrow[8], lrow[8];
    for (int v = 0; v < 8; ++v) { mrow[v] = -1e30f; lrow[v] = 0.f; }

    for (int s0 = 0; s0 < row0 + 16; s0 += 32) {
        // S = Q K^T (two 16x16 n-tiles, K-dim = 64)
        v8f sA = z, sB = z;
        {
            v16bf b0 = load_frag_B(kb, s0,      HDc, 0,  lane);
            sA = WMMA_BF16(qf0, b0, sA);
            v16bf b1 = load_frag_B(kb, s0 + 16, HDc, 0,  lane);
            sB = WMMA_BF16(qf0, b1, sB);
            v16bf b2 = load_frag_B(kb, s0,      HDc, 32, lane);
            sA = WMMA_BF16(qf1, b2, sA);
            v16bf b3 = load_frag_B(kb, s0 + 16, HDc, 32, lane);
            sB = WMMA_BF16(qf1, b3, sB);
        }

        // causal mask where the chunk overlaps the diagonal
        if (s0 + 31 > row0) {
            for (int v = 0; v < 8; ++v) {
                int r = row0 + v + 8 * hi;
                if (s0 + nlo      > r) sA[v] = -1e30f;
                if (s0 + 16 + nlo > r) sB[v] = -1e30f;
            }
        }

        // online softmax, per row (per vgpr index, 16-lane half reduction)
        for (int v = 0; v < 8; ++v) {
            float r = fmaxf(sA[v], sB[v]);
            r = fmaxf(r, __shfl_xor(r, 1, 32));
            r = fmaxf(r, __shfl_xor(r, 2, 32));
            r = fmaxf(r, __shfl_xor(r, 4, 32));
            r = fmaxf(r, __shfl_xor(r, 8, 32));
            float nm    = fmaxf(mrow[v], r);
            float alpha = __expf(mrow[v] - nm);
            mrow[v] = nm;
            float e0 = __expf(sA[v] - nm);
            float e1 = __expf(sB[v] - nm);
            sA[v] = e0; sB[v] = e1;
            float rs = e0 + e1;
            rs += __shfl_xor(rs, 1, 32);
            rs += __shfl_xor(rs, 2, 32);
            rs += __shfl_xor(rs, 4, 32);
            rs += __shfl_xor(rs, 8, 32);
            lrow[v] = lrow[v] * alpha + rs;
            for (int j = 0; j < 4; ++j) O[j][v] *= alpha;
        }

        // stage P (D-layout) into LDS row-major 16x32, reload as A-fragment
        for (int v = 0; v < 8; ++v) {
            int r = v + 8 * hi;
            lds[r * 32 + nlo]      = f2bf(sA[v]);
            lds[r * 32 + 16 + nlo] = f2bf(sB[v]);
        }
        v16bf ap;
        {
            const unsigned short* p = lds + (lane & 15) * 32 + hi * 8;
            V16U u;
            u.h[0] = *reinterpret_cast<const v8bf*>(p);
            u.h[1] = *reinterpret_cast<const v8bf*>(p + 16);
            ap = u.v;
        }

        // O += P * V   (B from V^T: row = d, K = s contiguous)
        for (int j = 0; j < 4; ++j) {
            v16bf bv = load_frag_B(vb, j * 16, Tc, s0, lane);
            O[j] = WMMA_BF16(ap, bv, O[j]);
        }
    }

    // normalize and store head-concatenated output [B*T, H*HD] bf16
    int b = bh >> 4, h = bh & 15;
    for (int v = 0; v < 8; ++v) {
        float inv = 1.0f / lrow[v];
        int t = row0 + v + 8 * hi;
        size_t rowg = (size_t)(b * Tc + t) * Cc;
        for (int j = 0; j < 4; ++j) {
            int col = h * HDc + j * 16 + nlo;
            attn[rowg + col] = f2bf(O[j][v] * inv);
        }
    }
}

// ---------------------------------------------------------------------------
// Output projection: [4096,1024] x [1024,1024] + bias -> f32
// ---------------------------------------------------------------------------
__global__ __launch_bounds__(128) void proj_kernel(
    const unsigned short* __restrict__ attn,
    const unsigned short* __restrict__ WpT,
    const float* __restrict__ bp,
    float* __restrict__ out) {
    int lane = threadIdx.x & 31;
    int w    = blockIdx.x * 4 + (threadIdx.x >> 5);   // 0..4095
    int nt   = w >> 8;                                // 16 col-tiles of 64
    int mt   = w & 255;
    int row0 = mt * 16, n0 = nt * 64;

    v8f z = {0.f, 0.f, 0.f, 0.f, 0.f, 0.f, 0.f, 0.f};
    v8f acc[4];
    for (int j = 0; j < 4; ++j) acc[j] = z;

    for (int k0 = 0; k0 < Cc; k0 += 32) {
        v16bf a = load_frag_A(attn, row0, Cc, k0, lane);
        for (int j = 0; j < 4; ++j) {
            v16bf b = load_frag_B(WpT, n0 + j * 16, Cc, k0, lane);
            acc[j] = WMMA_BF16(a, b, acc[j]);
        }
    }

    int hi = lane >> 4, nlo = lane & 15;
    for (int j = 0; j < 4; ++j) {
        for (int v = 0; v < 8; ++v) {
            int r   = row0 + v + 8 * hi;
            int col = n0 + j * 16 + nlo;
            out[(size_t)r * Cc + col] = acc[j][v] + bp[col];
        }
    }
}

// ---------------------------------------------------------------------------
extern "C" void kernel_launch(void* const* d_in, const int* in_sizes, int n_in,
                              void* d_out, int out_size, void* d_ws, size_t ws_size,
                              hipStream_t stream) {
    const float* x  = (const float*)d_in[0];
    const float* Wq = (const float*)d_in[1];
    const float* Wk = (const float*)d_in[2];
    const float* Wv = (const float*)d_in[3];
    const float* Wp = (const float*)d_in[4];
    const float* bp = (const float*)d_in[5];
    float* out = (float*)d_out;

    char* ws = (char*)d_ws;
    size_t off = 0;
    auto alloc = [&](size_t bytes) -> void* {
        void* p = ws + off;
        off = (off + bytes + 255) & ~(size_t)255;
        return p;
    };
    unsigned short* xb   = (unsigned short*)alloc((size_t)ROWS * Cc * 2);       // 8 MB
    unsigned short* WqT  = (unsigned short*)alloc((size_t)Hc * HDc * Cc * 2);   // 2 MB
    unsigned short* WkT  = (unsigned short*)alloc((size_t)Hc * HDc * Cc * 2);
    unsigned short* WvT  = (unsigned short*)alloc((size_t)Hc * HDc * Cc * 2);
    unsigned short* WpT  = (unsigned short*)alloc((size_t)Cc * Cc * 2);         // 2 MB
    unsigned short* Qb   = (unsigned short*)alloc((size_t)Bc * Hc * Tc * HDc * 2); // 8 MB
    unsigned short* Kb   = (unsigned short*)alloc((size_t)Bc * Hc * Tc * HDc * 2);
    unsigned short* VTb  = (unsigned short*)alloc((size_t)Bc * Hc * HDc * Tc * 2);
    unsigned short* attn = (unsigned short*)alloc((size_t)ROWS * Cc * 2);       // 8 MB

    // layout conversions
    {
        int n = ROWS * Cc;   // 4194304
        cvt_bf16_kernel<<<(n + 255) / 256, 256, 0, stream>>>(x, xb, n);
    }
    cvt_wT_kernel<<<(Hc * Cc * HDc) / 256, 256, 0, stream>>>(Wq, WqT);
    cvt_wT_kernel<<<(Hc * Cc * HDc) / 256, 256, 0, stream>>>(Wk, WkT);
    cvt_wT_kernel<<<(Hc * Cc * HDc) / 256, 256, 0, stream>>>(Wv, WvT);
    cvt_wpT_kernel<<<(Cc * Cc) / 256, 256, 0, stream>>>(Wp, WpT);

    // QKV projection: 4096 waves (256 m-tiles x 16 heads), 4 waves/block
    qkv_kernel<<<1024, 128, 0, stream>>>(xb, WqT, WkT, WvT, Qb, Kb, VTb);

    // flash attention: 4096 waves (128 t-tiles x 32 bh), 4 waves/block
    flash_kernel<<<1024, 128, 0, stream>>>(Qb, Kb, VTb, attn);

    // output projection: 4096 waves (256 m-tiles x 16 n-tiles)
    proj_kernel<<<1024, 128, 0, stream>>>(attn, WpT, bp, out);
}
